// Attn_head_41283225649258
// MI455X (gfx1250) — matrix operand order repaired and checked by reading, not compiled, not run on hardware
//
#include <hip/hip_runtime.h>
#include <hip/hip_bf16.h>

typedef __attribute__((ext_vector_type(16))) _Float16 v16h;
typedef __attribute__((ext_vector_type(8)))  _Float16 v8h;
typedef __attribute__((ext_vector_type(8)))  float    v8f;

#define N_NODES 8192
#define C_IN    512
#define F_OUT   256
#define NW      256            // mask words per row = N_NODES/32
#define NCH     (N_NODES / 32) // K-chunks in the attention GEMM

// =====================================================================
// Kernel 1: seq_fts_h = f16( W1 (256x512) * x (512x8192) )
// Block: 256 thr (8 waves, each wave = 16 f-rows x 128 n-cols)
// Grid: (64 n-blocks, 2 f-blocks)
// x-tile staged in LDS transposed [n=128][k=32], pad stride 40 halves
// (rows 16B-aligned, bank-stride-4 -> conflict-free v8h reads).
// B-fragment loads are 2-deep software-pipelined against the WMMAs so
// the ds_load -> wmma dependency is one tile ahead (no per-WMMA stall).
// =====================================================================
__global__ __launch_bounds__(256) void k_gemm1(const float* __restrict__ x,
                                               const float* __restrict__ W1,
                                               _Float16* __restrict__ seqh)
{
    __shared__ _Float16 ldsx[128 * 40];
    const int t    = threadIdx.x;
    const int lane = t & 31;
    const int wv   = t >> 5;
    const int nb   = blockIdx.x * 128;
    const int f0   = blockIdx.y * 128 + wv * 16;
    const int l15  = lane & 15;
    const bool lo16 = (lane < 16);
    const int cShift = lo16 ? 0 : 8;        // ISA 16-bit A layout lane split
    const float* wrow = W1 + (size_t)(f0 + l15) * C_IN;

    v8f acc[8];
    #pragma unroll
    for (int i = 0; i < 8; ++i)
        #pragma unroll
        for (int e = 0; e < 8; ++e) acc[i][e] = 0.0f;

    const int nn  = t & 127;
    const int cc2 = t >> 7;                 // 0..1
    const int koff = lo16 ? 0 : 8;          // B layout: lo lanes K0..7,16..23, hi K8..15,24..31

    auto loadB = [&](int tn) -> v16h {
        int n = tn * 16 + l15;
        v8h blo = *(const v8h*)&ldsx[n * 40 + koff];
        v8h bhi = *(const v8h*)&ldsx[n * 40 + koff + 16];
        return __builtin_shufflevector(blo, bhi,
                 0,1,2,3,4,5,6,7,8,9,10,11,12,13,14,15);
    };

    for (int c0 = 0; c0 < C_IN; c0 += 32) {
        // Stage x tile: coalesced f32 reads along n, f16 transposed store.
        #pragma unroll
        for (int p = 0; p < 16; ++p) {
            int cc = p * 2 + cc2;
            ldsx[nn * 40 + cc] = (_Float16)x[(size_t)(c0 + cc) * N_NODES + nb + nn];
        }
        __syncthreads();

        // A fragment from W1 (tiny, cache-resident), per ISA 16x32 layout.
        v16h a;
        #pragma unroll
        for (int e = 0; e < 8; ++e) {
            a[e]     = (_Float16)wrow[c0 + cShift + e];        // K 0..7 / 8..15
            a[e + 8] = (_Float16)wrow[c0 + 16 + cShift + e];   // K 16..23 / 24..31
        }

        // 2-deep pipeline: load tile tn+1 while WMMA consumes tile tn.
        v16h bcur = loadB(0);
        #pragma unroll
        for (int tn = 0; tn < 8; ++tn) {
            v16h bnxt = bcur;
            if (tn < 7) bnxt = loadB(tn + 1);
            acc[tn] = __builtin_amdgcn_wmma_f32_16x16x32_f16(
                        false, a, false, bcur, (short)0, acc[tn], false, false);
            bcur = bnxt;
        }
        __syncthreads();
    }

    const int mAdd = lo16 ? 0 : 8;
    #pragma unroll
    for (int tn = 0; tn < 8; ++tn) {
        int n = nb + tn * 16 + l15;
        #pragma unroll
        for (int r = 0; r < 8; ++r) {
            int f = f0 + mAdd + r;
            seqh[(size_t)f * N_NODES + n] = (_Float16)acc[tn][r];
        }
    }
}

// =====================================================================
// Kernel 2: f1[j], f2[j] from f16 seq_fts (coalesced column reads).
// =====================================================================
__global__ __launch_bounds__(256) void k_fvec(const _Float16* __restrict__ seqh,
                                              const float* __restrict__ w21,
                                              const float* __restrict__ b21,
                                              const float* __restrict__ w22,
                                              const float* __restrict__ b22,
                                              float* __restrict__ f1,
                                              float* __restrict__ f2)
{
    int j = blockIdx.x * 256 + threadIdx.x;
    float s1 = 0.f, s2 = 0.f;
    for (int f = 0; f < F_OUT; ++f) {
        float v = (float)seqh[(size_t)f * N_NODES + j];
        s1 += v * w21[f];
        s2 += v * w22[f];
    }
    f1[j] = s1 + b21[0];
    f2[j] = s2 + b22[0];
}

// =====================================================================
// Kernel 3: the single full pass over adj (268 MB = the HBM floor).
// Produces (a) deterministic 16-way partial softmax denominators and
// (b) a packed adjacency bitmask (8 MB) so the GEMM pass never rereads
// adj. Wave32 lanes = 32 consecutive j -> one ballot_w32 per row gives
// exactly one mask word; lane 0 stores it.
// Grid 512 = 32 j-blocks x 16 i-blocks; adj reads fully coalesced.
// =====================================================================
__global__ __launch_bounds__(256) void k_denom(const int* __restrict__ adj,
                                               const float* __restrict__ f1,
                                               const float* __restrict__ f2,
                                               float* __restrict__ part,
                                               unsigned* __restrict__ mask)
{
    int t  = threadIdx.x;
    int jb = blockIdx.x & 31;
    int ib = blockIdx.x >> 5;
    int j  = jb * 256 + t;
    int jw = j >> 5;                       // wave-uniform word index
    float f1j = f1[j];
    float sum = 0.f;
    int ibase = ib * 512;
    for (int r = 0; r < 512; ++r) {
        int i = ibase + r;
        int a = adj[(size_t)i * N_NODES + j];
        unsigned bal = __builtin_amdgcn_ballot_w32(a > 0);
        if ((t & 31) == 0)
            mask[(size_t)i * NW + jw] = bal;
        float v = f1j + f2[i];             // f2[i] uniform -> scalar load
        v = (v >= 0.f) ? v : 0.01f * v;    // LeakyReLU after mask-add
        sum += (a > 0) ? __expf(v) : 0.0f; // masked entries: exp(-1e7)=0
    }
    part[(size_t)ib * N_NODES + j] = sum;
}

// Fixed-order reduction of partials + reciprocal (deterministic).
__global__ __launch_bounds__(256) void k_recip(const float* __restrict__ part,
                                               float* __restrict__ rden)
{
    int j = blockIdx.x * 256 + threadIdx.x;
    float s = 0.f;
    #pragma unroll
    for (int ib = 0; ib < 16; ++ib) s += part[(size_t)ib * N_NODES + j];
    rden[j] = (s > 0.f) ? (1.0f / s) : 0.0f;
}

// =====================================================================
// Kernel 4: out[f][i] = elu( sum_j p[i,j]*seq_fts[f,j] + bias[f] )
// p regenerated on the fly from the PACKED mask + f1/f2/rden
// (flash-style; NxN never touches HBM; 8 MB mask is L2-resident).
// Block: 256 thr (8 waves = 2 i-groups x 4 f-groups). WG tile: 32i x 256f.
// Pipeline per chunk (double-buffered LDS p-tile, ONE barrier/chunk):
//   issue global B loads (chunk c) -> produce p-tile (chunk c+1, other
//   buffer, exp-heavy VALU overlaps the B loads) -> ds A-frag reads ->
//   4x WMMA -> s_barrier.
// =====================================================================
__global__ __launch_bounds__(256) void k_attn(const unsigned* __restrict__ mask,
                                              const float* __restrict__ f1,
                                              const float* __restrict__ f2,
                                              const float* __restrict__ rden,
                                              const _Float16* __restrict__ seqh,
                                              const float* __restrict__ bias,
                                              float* __restrict__ out)
{
    __shared__ _Float16 ldsp[2][32 * 40];
    const int t    = threadIdx.x;
    const int lane = t & 31;
    const int wv   = t >> 5;
    const int i0   = blockIdx.x * 32;
    const int iSub = (wv >> 2) * 16;       // 0 or 16
    const int fSub = (wv & 3) * 64;        // 0,64,128,192
    const int l15  = lane & 15;
    const bool lo16 = (lane < 16);

    // p-tile producer coords: thread -> (i = i0 + it*8 + iOff, j = j0 + jLane)
    const int jLane = t & 31;
    const int iOff  = t >> 5;              // 0..7
    float f2r[4];
    #pragma unroll
    for (int it = 0; it < 4; ++it) f2r[it] = f2[i0 + it * 8 + iOff];

    v8f acc[4];
    #pragma unroll
    for (int i = 0; i < 4; ++i)
        #pragma unroll
        for (int e = 0; e < 8; ++e) acc[i][e] = 0.0f;

    const int cShift = lo16 ? 0 : 8;
    const int rowI   = iSub + l15;

    auto produce = [&](int j0, int buf) {
        int   jc   = j0 + jLane;
        int   jw   = j0 >> 5;
        float f1j  = f1[jc];
        float rinv = rden[jc];
        #pragma unroll
        for (int it = 0; it < 4; ++it) {
            int i = i0 + it * 8 + iOff;
            unsigned m = mask[(size_t)i * NW + jw];   // wave-uniform word
            float v = f1j + f2r[it];
            v = (v >= 0.f) ? v : 0.01f * v;
            float p = ((m >> jLane) & 1u) ? __expf(v) * rinv : 0.0f;
            ldsp[buf][(it * 8 + iOff) * 40 + jLane] = (_Float16)p;
        }
        // prefetch next chunk's mask word (global_prefetch_b8)
        __builtin_prefetch(&mask[(size_t)(i0 + iOff) * NW + jw + 1], 0, 1);
    };

    produce(0, 0);
    __syncthreads();

    for (int c = 0; c < NCH; ++c) {
        const int j0  = c * 32;
        const int buf = c & 1;

        // Global B fragments for chunk c: independent of LDS state, issue
        // first so their (L2) latency overlaps the exp-heavy production.
        const int jsel = j0 + (lo16 ? 0 : 16);
        v16h bfr[4];
        #pragma unroll
        for (int tf = 0; tf < 4; ++tf) {
            int f = fSub + tf * 16 + l15;
            bfr[tf] = *(const v16h*)(seqh + (size_t)f * N_NODES + jsel);
        }

        // Produce next chunk's p-tile into the other buffer.
        if (c + 1 < NCH) produce(j0 + 32, buf ^ 1);

        // A fragment (16x32 f16) from this chunk's buffer, contiguous k = 16h + s, same order as B.
        v8h alo = *(const v8h*)&ldsp[buf][rowI * 40 + 2 * cShift];
        v8h ahi = *(const v8h*)&ldsp[buf][rowI * 40 + 2 * cShift + 8];
        v16h a  = __builtin_shufflevector(alo, ahi,
                    0,1,2,3,4,5,6,7,8,9,10,11,12,13,14,15);

        #pragma unroll
        for (int tf = 0; tf < 4; ++tf)
            acc[tf] = __builtin_amdgcn_wmma_f32_16x16x32_f16(
                        false, a, false, bfr[tf], (short)0, acc[tf], false, false);

        // One barrier per chunk: publishes produce(c+1), and protects buf
        // from being overwritten by produce(c+2) in the next iteration.
        __syncthreads();
    }

    // Epilogue: lane owns fixed f; 8 accum rows contiguous in i ->
    // bias + ELU, two aligned float4 stores per tile (transposed output).
    const int iBase = i0 + iSub + (lo16 ? 0 : 8);
    #pragma unroll
    for (int tf = 0; tf < 4; ++tf) {
        int f = fSub + tf * 16 + l15;
        float bf = bias[f];
        float o[8];
        #pragma unroll
        for (int r = 0; r < 8; ++r) {
            float v = acc[tf][r] + bf;
            o[r] = (v > 0.f) ? v : (__expf(v) - 1.0f);   // ELU
        }
        float* dst = out + (size_t)f * N_NODES + iBase;
        *(float4*)(dst)     = make_float4(o[0], o[1], o[2], o[3]);
        *(float4*)(dst + 4) = make_float4(o[4], o[5], o[6], o[7]);
    }
}

// =====================================================================
// Workspace layout (bytes):
//   [0,        4194304)  seq_fts f16 [256][8192]
//   [4194304,  +32768)   f1
//   [+32768,   +65536)   f2
//   [+65536,   +98304)   rden
//   [+98304,   +622592)  part [16][8192] f32
//   [4816896,  13205504) mask [8192][256] u32 (packed adjacency bits)
// Total ~12.6 MB.
// =====================================================================
extern "C" void kernel_launch(void* const* d_in, const int* in_sizes, int n_in,
                              void* d_out, int out_size, void* d_ws, size_t ws_size,
                              hipStream_t stream) {
    const float* x    = (const float*)d_in[0];
    const int*   adj  = (const int*)d_in[1];
    const float* W1   = (const float*)d_in[2];
    const float* w21  = (const float*)d_in[3];
    const float* b21  = (const float*)d_in[4];
    const float* w22  = (const float*)d_in[5];
    const float* b22  = (const float*)d_in[6];
    const float* bias = (const float*)d_in[7];
    float* out = (float*)d_out;

    char* ws = (char*)d_ws;
    _Float16* seqh = (_Float16*)ws;
    char* base2 = ws + (size_t)F_OUT * N_NODES * 2;   // 4,194,304
    float*    f1   = (float*)(base2);
    float*    f2   = (float*)(base2 + 32768);
    float*    rden = (float*)(base2 + 65536);
    float*    part = (float*)(base2 + 98304);
    unsigned* mask = (unsigned*)(base2 + 98304 + 524288);

    k_gemm1<<<dim3(64, 2), 256, 0, stream>>>(x, W1, seqh);
    k_fvec <<<dim3(32),    256, 0, stream>>>(seqh, w21, b21, w22, b22, f1, f2);
    k_denom<<<dim3(512),   256, 0, stream>>>(adj, f1, f2, part, mask);
    k_recip<<<dim3(32),    256, 0, stream>>>(part, rden);
    k_attn <<<dim3(256),   256, 0, stream>>>(mask, f1, f2, rden, seqh, bias, out);
}
